// SSENet_42932493090852
// MI455X (gfx1250) — matrix operand order repaired and checked by reading, not compiled, not run on hardware
//
#include <hip/hip_runtime.h>
#include <hip/hip_bf16.h>

// ---------------------------------------------------------------------------
// Problem constants (from reference)
// ---------------------------------------------------------------------------
#define HH      1024
#define POOL    8
#define PH      128          // pooled / feat spatial size
#define NF      16           // conv output channels
#define NS      50
#define KCONV   5
#define PADC    2
#define D_IN    69           // 2 + 1 + 50 + 16
#define D_PAD   96           // D_IN padded to multiple of 32
#define H1      128
#define H2      128
#define H3      96
#define EPSF    0.0001f

typedef __attribute__((ext_vector_type(16))) _Float16 v16h;
typedef __attribute__((ext_vector_type(8)))  float    v8f;

// ---------------------------------------------------------------------------
// WMMA helpers
// ---------------------------------------------------------------------------
__device__ __forceinline__ v8f wmma_f16(v16h a, v16h b, v8f c) {
  // (neg_a, A, neg_b, B, c_mod, C, reuse_a, reuse_b)
  return __builtin_amdgcn_wmma_f32_16x16x32_f16(false, a, false, b, (short)0, c,
                                                false, false);
}

// Build a 16x32 f16 A-fragment for this lane from an LDS row.
// lanerow = pointer to this lane's row (row = lane&15 within the wave tile).
// ISA layout (16-bit A, 16x32): lanes 0-15: halves 0-7 -> K 0-7, halves 8-15 -> K 16-23
//                               lanes 16-31: halves 0-7 -> K 8-15, halves 8-15 -> K 24-31
__device__ __forceinline__ v16h load_A(const _Float16* lanerow, int hi, int kbase) {
  union { v16h v; unsigned int u[8]; } r;
#pragma unroll
  for (int j = 0; j < 8; ++j) {
    int K = kbase + 2 * j + 8 * hi + ((j >= 4) ? 8 : 0);   // even -> dword aligned
    r.u[j] = *(const unsigned int*)(lanerow + K);
  }
  return r.v;
}

// B fragment: pre-swizzled in workspace, 32 lanes x 16 halves contiguous per frag.
__device__ __forceinline__ v16h load_B(const _Float16* packed, int frag, int ln) {
  return *(const v16h*)(packed + ((size_t)(frag * 32 + ln) << 4));
}

// ---------------------------------------------------------------------------
// Kernel 1: 8x8 average pool  sdf(2,1024,1024) -> pooled(2,128,128)
// ---------------------------------------------------------------------------
__global__ void pool_kernel(const float* __restrict__ sdf, float* __restrict__ pooled) {
  int idx = blockIdx.x * blockDim.x + threadIdx.x;
  if (idx >= 2 * PH * PH) return;
  int c = idx / (PH * PH);
  int rem = idx - c * PH * PH;
  int y = rem / PH, x = rem - y * PH;
  const float* base = sdf + (size_t)c * HH * HH + (size_t)(y * POOL) * HH + x * POOL;
  float s = 0.f;
#pragma unroll
  for (int i = 0; i < POOL; ++i)
#pragma unroll
    for (int j = 0; j < POOL; ++j) s += base[i * HH + j];
  pooled[idx] = s * (1.0f / 64.0f);
}

// ---------------------------------------------------------------------------
// Kernel 2: 5x5 conv, 2 -> 16 channels, pad 2, + bias  -> feat(16,128,128)
// ---------------------------------------------------------------------------
__global__ void conv_kernel(const float* __restrict__ pooled,
                            const float* __restrict__ cw,
                            const float* __restrict__ cb,
                            float* __restrict__ feat) {
  int idx = blockIdx.x * blockDim.x + threadIdx.x;
  if (idx >= NF * PH * PH) return;
  int o = idx / (PH * PH);
  int rem = idx - o * PH * PH;
  int y = rem / PH, x = rem - y * PH;
  float acc = cb[o];
#pragma unroll
  for (int i = 0; i < 2; ++i) {
    const float* pc = pooled + i * PH * PH;
    const float* wc = cw + ((o * 2 + i) * KCONV) * KCONV;
#pragma unroll
    for (int ky = 0; ky < KCONV; ++ky) {
      int yy = y + ky - PADC;
      if (yy < 0 || yy >= PH) continue;
#pragma unroll
      for (int kx = 0; kx < KCONV; ++kx) {
        int xx = x + kx - PADC;
        if (xx < 0 || xx >= PH) continue;
        acc += pc[yy * PH + xx] * wc[ky * KCONV + kx];
      }
    }
  }
  feat[idx] = acc;
}

// ---------------------------------------------------------------------------
// Kernel 3: pack fp32 weights (K x N row-major) into f16 B-fragments.
// B layout (16-bit, 32x16 per frag): h<8 -> K=8*(lane>>4)+h, h>=8 -> K=16+8*(lane>>4)+h-8,
// N = lane&15.  frag index = kt*ntiles + nt.
// ---------------------------------------------------------------------------
__global__ void pack_weights(const float* __restrict__ W, int K, int Nn,
                             int ntiles, int total, _Float16* __restrict__ outp) {
  int e = blockIdx.x * blockDim.x + threadIdx.x;
  if (e >= total) return;
  int frag = e >> 9;            // 512 halves per fragment
  int within = e & 511;
  int ln = within >> 4;
  int h = within & 15;
  int kt = frag / ntiles;
  int nt = frag - kt * ntiles;
  int k = kt * 32 + ((h < 8) ? (8 * (ln >> 4) + h) : (16 + 8 * (ln >> 4) + h - 8));
  int n = nt * 16 + (ln & 15);
  float v = (k < K && n < Nn) ? W[(size_t)k * Nn + n] : 0.0f;
  outp[e] = (_Float16)v;
}

// ---------------------------------------------------------------------------
// Kernel 4: fused interp + 4-layer WMMA MLP.
// One block = 256 threads = 8 waves = 128 points; each wave owns 16 points.
// ---------------------------------------------------------------------------
__global__ void __launch_bounds__(256, 1)
mlp_kernel(const float* __restrict__ x, const float* __restrict__ s,
           const float* __restrict__ sse, const float* __restrict__ feat,
           const float* __restrict__ b1, const float* __restrict__ b2,
           const float* __restrict__ b3, const float* __restrict__ b4,
           const _Float16* __restrict__ w1p, const _Float16* __restrict__ w2p,
           const _Float16* __restrict__ w3p, const _Float16* __restrict__ w4p,
           float* __restrict__ out, int N) {
  __shared__ _Float16 act[128 * D_PAD];      // 24 KB: input activations (f16)
  __shared__ _Float16 hbuf[8 * 16 * 128];    // 32 KB: per-wave hidden activations
  __shared__ int   pix0[128], piy0[128];
  __shared__ float pwx[128], pwy[128];

  const int tid = threadIdx.x;
  const int base_row = blockIdx.x * 128;

  // ---- stage 1a: coords + raw scalar features + K-padding --------------------
  for (int p = tid; p < 128; p += 256) {
    int row = base_row + p;
    if (row >= N) row = N - 1;
    float fx = x[2 * row], fy = x[2 * row + 1];
    float gx = fminf(fmaxf(fx, EPSF), 1.0f - EPSF) * (float)(PH - 1);
    float gy = fminf(fmaxf(fy, EPSF), 1.0f - EPSF) * (float)(PH - 1);
    int x0 = (int)floorf(gx);
    int y0 = (int)floorf(gy);
    pix0[p] = x0; piy0[p] = y0;
    pwx[p] = gx - (float)x0; pwy[p] = gy - (float)y0;
    _Float16* ar = act + p * D_PAD;
    ar[0] = (_Float16)fx;
    ar[1] = (_Float16)fy;
    ar[2] = (_Float16)s[row];
#pragma unroll
    for (int c = D_IN; c < D_PAD; ++c) ar[c] = (_Float16)0.0f;
  }
  // ---- stage 1b: coalesced sse load (addresses are fully contiguous) ---------
  {
    int maxrow = N - 1;
    for (int idx = tid; idx < 128 * NS; idx += 256) {
      int p = idx / NS, c = idx - p * NS;
      int row = base_row + p; if (row > maxrow) row = maxrow;
      act[p * D_PAD + 3 + c] = (_Float16)sse[(size_t)row * NS + c];
    }
  }
  __syncthreads();
  // ---- stage 1c: bilinear interp of 16 feature channels ----------------------
  for (int idx = tid; idx < 128 * NF; idx += 256) {
    int p = idx & 127;
    int c = idx >> 7;
    int x0 = pix0[p], y0 = piy0[p];
    float wx = pwx[p], wy = pwy[p];
    const float* Fc = feat + (size_t)c * PH * PH;
    float f00 = Fc[y0 * PH + x0];
    float f10 = Fc[y0 * PH + x0 + 1];
    float f01 = Fc[(y0 + 1) * PH + x0];
    float f11 = Fc[(y0 + 1) * PH + x0 + 1];
    float v = f00 * (1.f - wx) * (1.f - wy) + f10 * wx * (1.f - wy) +
              f01 * (1.f - wx) * wy + f11 * wx * wy;
    act[p * D_PAD + 53 + c] = (_Float16)v;
  }
  __syncthreads();

  // ---- stage 2: per-wave 4-layer WMMA MLP on its 16-row tile -----------------
  const int wv = tid >> 5;
  const int ln = tid & 31;
  const int rl = ln & 15;        // row within tile / output column
  const int hi = ln >> 4;        // lane half select
  _Float16* hb = hbuf + wv * (16 * 128);

  // Layer 1: 96(pad of 69) -> 128
  {
    const _Float16* lanerow = act + (wv * 16 + rl) * D_PAD;
    v16h a0 = load_A(lanerow, hi, 0);
    v16h a1 = load_A(lanerow, hi, 32);
    v16h a2 = load_A(lanerow, hi, 64);
#pragma unroll
    for (int nt = 0; nt < 8; ++nt) {
      v8f c = {};
      c = wmma_f16(a0, load_B(w1p, 0 * 8 + nt, ln), c);
      c = wmma_f16(a1, load_B(w1p, 1 * 8 + nt, ln), c);
      c = wmma_f16(a2, load_B(w1p, 2 * 8 + nt, ln), c);
      int n = nt * 16 + rl;
      float bias = b1[n];
#pragma unroll
      for (int r = 0; r < 8; ++r) {
        float v = fmaxf(c[r] + bias, 0.0f);
        hb[(r + 8 * hi) * 128 + n] = (_Float16)v;
      }
    }
  }
  // Layer 2: 128 -> 128
  {
    const _Float16* lanerow = hb + rl * 128;
    v16h a[4];
#pragma unroll
    for (int kt = 0; kt < 4; ++kt) a[kt] = load_A(lanerow, hi, kt * 32);
#pragma unroll
    for (int nt = 0; nt < 8; ++nt) {
      v8f c = {};
#pragma unroll
      for (int kt = 0; kt < 4; ++kt)
        c = wmma_f16(a[kt], load_B(w2p, kt * 8 + nt, ln), c);
      int n = nt * 16 + rl;
      float bias = b2[n];
#pragma unroll
      for (int r = 0; r < 8; ++r) {
        float v = fmaxf(c[r] + bias, 0.0f);
        hb[(r + 8 * hi) * 128 + n] = (_Float16)v;
      }
    }
  }
  // Layer 3: 128 -> 96 (output stored with stride 96)
  {
    const _Float16* lanerow = hb + rl * 128;
    v16h a[4];
#pragma unroll
    for (int kt = 0; kt < 4; ++kt) a[kt] = load_A(lanerow, hi, kt * 32);
#pragma unroll
    for (int nt = 0; nt < 6; ++nt) {
      v8f c = {};
#pragma unroll
      for (int kt = 0; kt < 4; ++kt)
        c = wmma_f16(a[kt], load_B(w3p, kt * 6 + nt, ln), c);
      int n = nt * 16 + rl;
      float bias = b3[n];
#pragma unroll
      for (int r = 0; r < 8; ++r) {
        float v = fmaxf(c[r] + bias, 0.0f);
        hb[(r + 8 * hi) * 96 + n] = (_Float16)v;
      }
    }
  }
  // Layer 4: 96 -> 1 (N padded to 16; only column 0 is real)
  {
    const _Float16* lanerow = hb + rl * 96;
    v8f c = {};
#pragma unroll
    for (int kt = 0; kt < 3; ++kt)
      c = wmma_f16(load_A(lanerow, hi, kt * 32), load_B(w4p, kt, ln), c);
    float bias = b4[0];
    if (rl == 0) {
#pragma unroll
      for (int r = 0; r < 8; ++r) {
        int row = base_row + wv * 16 + r + 8 * hi;
        if (row < N) out[row] = c[r] + bias;
      }
    }
  }
}

// ---------------------------------------------------------------------------
// Launcher
// ---------------------------------------------------------------------------
extern "C" void kernel_launch(void* const* d_in, const int* in_sizes, int n_in,
                              void* d_out, int out_size, void* d_ws, size_t ws_size,
                              hipStream_t stream) {
  const float* x      = (const float*)d_in[0];
  const float* s      = (const float*)d_in[1];
  const float* sse    = (const float*)d_in[2];
  const float* sdf    = (const float*)d_in[3];
  const float* conv_w = (const float*)d_in[4];
  const float* conv_b = (const float*)d_in[5];
  const float* w1     = (const float*)d_in[6];
  const float* b1     = (const float*)d_in[7];
  const float* w2     = (const float*)d_in[8];
  const float* b2     = (const float*)d_in[9];
  const float* w3     = (const float*)d_in[10];
  const float* b3     = (const float*)d_in[11];
  const float* w4     = (const float*)d_in[12];
  const float* b4     = (const float*)d_in[13];
  float* out = (float*)d_out;
  const int N = in_sizes[0] / 2;

  // workspace layout
  float* pooled = (float*)d_ws;                       //  2*128*128 f32
  float* feat   = pooled + 2 * PH * PH;               // 16*128*128 f32
  _Float16* w1p = (_Float16*)(feat + NF * PH * PH);   // 24 frags
  _Float16* w2p = w1p + 24 * 512;                     // 32 frags
  _Float16* w3p = w2p + 32 * 512;                     // 24 frags
  _Float16* w4p = w3p + 24 * 512;                     //  3 frags

  pool_kernel<<<(2 * PH * PH + 255) / 256, 256, 0, stream>>>(sdf, pooled);
  conv_kernel<<<(NF * PH * PH + 255) / 256, 256, 0, stream>>>(pooled, conv_w,
                                                              conv_b, feat);
  pack_weights<<<(24 * 512 + 255) / 256, 256, 0, stream>>>(w1, D_IN, H1, 8,
                                                           24 * 512, w1p);
  pack_weights<<<(32 * 512 + 255) / 256, 256, 0, stream>>>(w2, H1, H2, 8,
                                                           32 * 512, w2p);
  pack_weights<<<(24 * 512 + 255) / 256, 256, 0, stream>>>(w3, H2, H3, 6,
                                                           24 * 512, w3p);
  pack_weights<<<(3 * 512 + 255) / 256, 256, 0, stream>>>(w4, H3, 1, 1,
                                                          3 * 512, w4p);

  int blocks = (N + 127) / 128;
  mlp_kernel<<<blocks, 256, 0, stream>>>(x, s, sse, feat, b1, b2, b3, b4,
                                         w1p, w2p, w3p, w4p, out, N);
}
